// WindowAttention_22505628631715
// MI455X (gfx1250) — matrix operand order repaired and checked by reading, not compiled, not run on hardware
//
#include <hip/hip_runtime.h>

#define WSZ 7
#define L   49
#define LP  64
#define C   256
#define H   8
#define DH  32
#define C3  768
#define NW  1024
#define NBLK 8192

// ---- LDS layout (in 16-bit units) ----
#define XS_OFF 0              // x tile, 64 x 264 (bf16)           = 16896 shorts
#define QK_OFF 16896          // per head: q 64x40, k 64x40        = 8*5120 shorts
#define VT_OFF 57856          // per head: vT 32x72                = 8*2304 shorts
#define LDS_SHORTS 76288      // 152576 bytes

// ---- ws layout (bytes) ----
#define WQKV_OFF  0           // bf16 [768][256]  = 393216 B
#define WPROJ_OFF 393216      // bf16 [256][256]  = 131072 B
#define BIAS_OFF  524288      // f32  [8][64][64] = 131072 B
#define QB_OFF    655360      // f32  [768]       =   3072 B

typedef __attribute__((ext_vector_type(16))) __bf16          v16bf;
typedef __attribute__((ext_vector_type(8)))  float           v8f;
typedef __attribute__((ext_vector_type(8)))  unsigned short  u16x8;

union Frag16 { v16bf bf; u16x8 us[2]; };

__device__ __forceinline__ unsigned short f2bf(float f) {
  unsigned int u = __float_as_uint(f);
  u += 0x7FFFu + ((u >> 16) & 1u);          // round-to-nearest-even
  return (unsigned short)(u >> 16);
}

// ---------------------------------------------------------------------------
// Prep: bf16 transposed weights (scale folded into Q), padded rel-bias table,
// scaled qkv bias.
// ---------------------------------------------------------------------------
__global__ __launch_bounds__(256) void wa_prep_kernel(
    const float* __restrict__ qkv_w, const float* __restrict__ qkv_b,
    const float* __restrict__ rel_bias, const float* __restrict__ proj_w,
    unsigned short* __restrict__ wqkvT, unsigned short* __restrict__ wprojT,
    float* __restrict__ biasp, float* __restrict__ qbs) {
  const int i = blockIdx.x * blockDim.x + threadIdx.x;
  const float scale = 0.17677669529663687f;  // 1/sqrt(32)

  if (i < C3 * C) {                       // qkv_wT[c][k] = qkv_w[k][c] (*scale if q)
    int c = i >> 8, k = i & 255;
    float v = qkv_w[k * C3 + c];
    if (c < C) v *= scale;
    wqkvT[c * C + k] = f2bf(v);
  }
  if (i < C * C) {                        // proj_wT[c][k] = proj_w[k][c]
    int c = i >> 8, k = i & 255;
    wprojT[c * C + k] = f2bf(proj_w[k * C + c]);
  }
  if (i < H * LP * LP) {                  // padded bias[h][m][n]
    int h = i >> 12, m = (i >> 6) & 63, n = i & 63;
    float v = 0.f;
    if (m < L && n < L) {
      int d0 = (m / WSZ) - (n / WSZ) + (WSZ - 1);
      int d1 = (m % WSZ) - (n % WSZ) + (WSZ - 1);
      v = rel_bias[(d0 * (2 * WSZ - 1) + d1) * H + h];
    }
    biasp[i] = v;
  }
  if (i < C3) {                           // scaled qkv bias
    float v = qkv_b[i];
    if (i < C) v *= scale;
    qbs[i] = v;
  }
}

// ---------------------------------------------------------------------------
// Fused window attention: one workgroup (8 waves) per window.
// ---------------------------------------------------------------------------
__global__ __launch_bounds__(256) void wa_attn_kernel(
    const float* __restrict__ x, const float* __restrict__ mask,
    const int* __restrict__ with_mask,
    const unsigned short* __restrict__ wqkvT,
    const unsigned short* __restrict__ wprojT,
    const float* __restrict__ biasp, const float* __restrict__ qbs,
    const float* __restrict__ proj_b, float* __restrict__ out) {
  extern __shared__ unsigned short lds[];
  const int b    = blockIdx.x;
  const int tid  = threadIdx.x;
  const int wave = tid >> 5;
  const int lane = tid & 31;
  const int l16  = lane & 15;
  const int half = lane >> 4;

  // ---- stage 0a: async DMA x[b] (49x256 f32 = 50176 B) into LDS staging
  //                (the q/k region, which is written only after this is dead).
  {
    const char* xb = (const char*)(x + (size_t)b * (L * C));
    // warm the mask window for this block while the async copies run
    const float* mkp = mask + (size_t)(b & (NW - 1)) * (L * L);
    __builtin_prefetch(mkp + tid * 16, 0, 1);
    for (int i = tid; i < (L * C) / 4; i += 256) {      // 3136 x 16B chunks
      unsigned int ldsa = (unsigned int)(QK_OFF * 2 + i * 16);
      const char* ga = xb + i * 16;
      asm volatile("global_load_async_to_lds_b128 %0, %1, off"
                   :: "v"(ldsa), "v"(ga) : "memory");
    }
    asm volatile("s_wait_asynccnt 0x0" ::: "memory");
  }
  __syncthreads();

  // ---- stage 0b: convert staged f32 -> bf16 x tile, zero pad rows ----
  {
    const float* xf = (const float*)((const char*)lds + QK_OFF * 2);
    for (int idx = tid; idx < L * C; idx += 256) {
      int m = idx >> 8, k = idx & 255;
      lds[XS_OFF + m * 264 + k] = f2bf(xf[idx]);
    }
    for (int idx = tid; idx < (LP - L) * C; idx += 256) {
      int m = L + (idx >> 8), k = idx & 255;
      lds[XS_OFF + m * 264 + k] = 0;
    }
  }
  __syncthreads();

  // ---- stage 1: QKV GEMM (64x256 @ 256x768 + bias) ----
  // q,k stored row-major [head][tok][dh]; v stored transposed [head][dh][tok].
  for (int nti = 0; nti < 6; ++nti) {
    const int nt = wave * 6 + nti;
    const int c  = nt * 16 + l16;            // output channel 0..767
    const float bv = qbs[c];
    for (int mt = 0; mt < 4; ++mt) {
      v8f acc = {bv, bv, bv, bv, bv, bv, bv, bv};
      const unsigned short* abase = &lds[XS_OFF + (mt * 16 + l16) * 264 + half * 8];
      Frag16 a_cur;
      a_cur.us[0] = *(const u16x8*)(abase);
      a_cur.us[1] = *(const u16x8*)(abase + 16);
#pragma unroll
      for (int kt = 0; kt < 8; ++kt) {
        Frag16 a_nxt;
        if (kt < 7) {                       // software pipeline: prefetch next A frag
          const unsigned short* ap = abase + (kt + 1) * 32;
          a_nxt.us[0] = *(const u16x8*)(ap);
          a_nxt.us[1] = *(const u16x8*)(ap + 16);
        }
        Frag16 bw;
        const unsigned short* bp = wqkvT + c * C + kt * 32 + half * 8;
        bw.us[0] = *(const u16x8*)(bp);
        bw.us[1] = *(const u16x8*)(bp + 16);
        acc = __builtin_amdgcn_wmma_f32_16x16x32_bf16(false, a_cur.bf, false, bw.bf,
                                                      (short)0, acc, false, false);
        if (kt < 7) a_cur = a_nxt;
      }
      const int hh = (c >> 5) & 7;
      const int dh = c & 31;
      const int tok0 = mt * 16 + half * 8;   // C-frag: M = r + 8*half
      if (c < 2 * C) {                        // q (part 0) / k (part 1): row-major
        const int part = c >> 8;
        unsigned short* dst = &lds[QK_OFF + hh * 5120 + part * 2560 + dh];
#pragma unroll
        for (int r = 0; r < 8; ++r) dst[(tok0 + r) * 40] = f2bf(acc[r]);
      } else {                                // v: transposed store, one b128
        u16x8 pk;
#pragma unroll
        for (int r = 0; r < 8; ++r) pk[r] = f2bf(acc[r]);
        *(u16x8*)(&lds[VT_OFF + hh * 2304 + dh * 72 + tok0]) = pk;
      }
    }
  }
  __syncthreads();

  // ---- stage 2: per-head attention (wave h owns head h) ----
  {
    const int h = wave;
    const unsigned short* qh = &lds[QK_OFF + h * 5120];
    const unsigned short* kh = qh + 2560;
    unsigned short* Ph = &lds[QK_OFF + h * 5120];   // overlays dead q/k region
    const float* bh = biasp + (h << 12);
    const int wm = with_mask[0];
    const float* mk = mask + (size_t)(b & (NW - 1)) * (L * L);

    // load all Q A-frags and K B-frags before overlaying q/k with P
    Frag16 aq[4], bk[4];
#pragma unroll
    for (int t = 0; t < 4; ++t) {
      const unsigned short* ap = qh + (t * 16 + l16) * 40 + half * 8;
      aq[t].us[0] = *(const u16x8*)(ap);
      aq[t].us[1] = *(const u16x8*)(ap + 16);
      const unsigned short* bp = kh + (t * 16 + l16) * 40 + half * 8;
      bk[t].us[0] = *(const u16x8*)(bp);
      bk[t].us[1] = *(const u16x8*)(bp + 16);
    }

    for (int mt = 0; mt < 4; ++mt) {
      v8f s[4];
#pragma unroll
      for (int nt = 0; nt < 4; ++nt) {
        v8f z = {0.f, 0.f, 0.f, 0.f, 0.f, 0.f, 0.f, 0.f};
        s[nt] = __builtin_amdgcn_wmma_f32_16x16x32_bf16(false, aq[mt].bf, false, bk[nt].bf,
                                                        (short)0, z, false, false);
      }
      // bias + mask + padding mask, then row softmax (row split across 16 lanes)
#pragma unroll
      for (int r = 0; r < 8; ++r) {
        const int m = mt * 16 + r + half * 8;
        float vals[4];
#pragma unroll
        for (int nt = 0; nt < 4; ++nt) {
          const int n = nt * 16 + l16;
          float v = s[nt][r];
          if (n < L) {
            v += bh[(m << 6) + n];
            if (wm && m < L) v += mk[m * L + n];
          } else {
            v = -1.0e30f;
          }
          vals[nt] = v;
        }
        float mx = fmaxf(fmaxf(vals[0], vals[1]), fmaxf(vals[2], vals[3]));
        mx = fmaxf(mx, __shfl_xor(mx, 1, 32));
        mx = fmaxf(mx, __shfl_xor(mx, 2, 32));
        mx = fmaxf(mx, __shfl_xor(mx, 4, 32));
        mx = fmaxf(mx, __shfl_xor(mx, 8, 32));
        float e0 = __expf(vals[0] - mx), e1 = __expf(vals[1] - mx);
        float e2 = __expf(vals[2] - mx), e3 = __expf(vals[3] - mx);
        float sm = (e0 + e1) + (e2 + e3);
        sm += __shfl_xor(sm, 1, 32);
        sm += __shfl_xor(sm, 2, 32);
        sm += __shfl_xor(sm, 4, 32);
        sm += __shfl_xor(sm, 8, 32);
        const float inv = 1.0f / sm;
        unsigned short* pr = Ph + m * 72;
        pr[l16]      = f2bf(e0 * inv);
        pr[16 + l16] = f2bf(e1 * inv);
        pr[32 + l16] = f2bf(e2 * inv);
        pr[48 + l16] = f2bf(e3 * inv);
      }
    }

    // O = P @ V  (M=64, N=32, K=64); store into O_all (overlays XS region)
    const unsigned short* vh = &lds[VT_OFF + h * 2304];
    for (int mt = 0; mt < 4; ++mt) {
      for (int nt = 0; nt < 2; ++nt) {
        v8f acc = {0.f, 0.f, 0.f, 0.f, 0.f, 0.f, 0.f, 0.f};
#pragma unroll
        for (int kt = 0; kt < 2; ++kt) {
          Frag16 pa, vb;
          const unsigned short* pp = Ph + (mt * 16 + l16) * 72 + kt * 32 + half * 8;
          pa.us[0] = *(const u16x8*)(pp);
          pa.us[1] = *(const u16x8*)(pp + 16);
          const unsigned short* vp = vh + (nt * 16 + l16) * 72 + kt * 32 + half * 8;
          vb.us[0] = *(const u16x8*)(vp);
          vb.us[1] = *(const u16x8*)(vp + 16);
          acc = __builtin_amdgcn_wmma_f32_16x16x32_bf16(false, pa.bf, false, vb.bf,
                                                        (short)0, acc, false, false);
        }
        const int cc = h * 32 + nt * 16 + l16;
        const int tok0 = mt * 16 + half * 8;
#pragma unroll
        for (int r = 0; r < 8; ++r)
          lds[XS_OFF + (tok0 + r) * 264 + cc] = f2bf(acc[r]);
      }
    }
  }
  __syncthreads();

  // ---- stage 3: output projection (64x256 @ 256x256 + bias) -> global ----
  for (int nti = 0; nti < 2; ++nti) {
    const int nt = wave * 2 + nti;
    const int cc = nt * 16 + l16;
    const float pb = proj_b[cc];
    for (int mt = 0; mt < 4; ++mt) {
      v8f acc = {pb, pb, pb, pb, pb, pb, pb, pb};
      const unsigned short* abase = &lds[XS_OFF + (mt * 16 + l16) * 264 + half * 8];
      Frag16 a_cur;
      a_cur.us[0] = *(const u16x8*)(abase);
      a_cur.us[1] = *(const u16x8*)(abase + 16);
#pragma unroll
      for (int kt = 0; kt < 8; ++kt) {
        Frag16 a_nxt;
        if (kt < 7) {
          const unsigned short* ap = abase + (kt + 1) * 32;
          a_nxt.us[0] = *(const u16x8*)(ap);
          a_nxt.us[1] = *(const u16x8*)(ap + 16);
        }
        Frag16 bw;
        const unsigned short* bp = wprojT + cc * C + kt * 32 + half * 8;
        bw.us[0] = *(const u16x8*)(bp);
        bw.us[1] = *(const u16x8*)(bp + 16);
        acc = __builtin_amdgcn_wmma_f32_16x16x32_bf16(false, a_cur.bf, false, bw.bf,
                                                      (short)0, acc, false, false);
        if (kt < 7) a_cur = a_nxt;
      }
      float* ob = out + (size_t)b * (L * C);
      const int tok0 = mt * 16 + half * 8;
#pragma unroll
      for (int r = 0; r < 8; ++r) {
        const int tok = tok0 + r;
        if (tok < L) ob[tok * C + cc] = acc[r];
      }
    }
  }
}

// ---------------------------------------------------------------------------
extern "C" void kernel_launch(void* const* d_in, const int* in_sizes, int n_in,
                              void* d_out, int out_size, void* d_ws, size_t ws_size,
                              hipStream_t stream) {
  const float* x        = (const float*)d_in[0];
  const float* mask     = (const float*)d_in[1];
  const float* qkv_w    = (const float*)d_in[2];
  const float* qkv_b    = (const float*)d_in[3];
  const float* rel_bias = (const float*)d_in[4];
  const float* proj_w   = (const float*)d_in[5];
  const float* proj_b   = (const float*)d_in[6];
  const int*   wm       = (const int*)d_in[7];
  float* out = (float*)d_out;

  char* ws = (char*)d_ws;
  unsigned short* wqkvT  = (unsigned short*)(ws + WQKV_OFF);
  unsigned short* wprojT = (unsigned short*)(ws + WPROJ_OFF);
  float* biasp = (float*)(ws + BIAS_OFF);
  float* qbs   = (float*)(ws + QB_OFF);

  wa_prep_kernel<<<dim3(768), dim3(256), 0, stream>>>(
      qkv_w, qkv_b, rel_bias, proj_w, wqkvT, wprojT, biasp, qbs);

  const size_t smem = (size_t)LDS_SHORTS * sizeof(unsigned short);  // 152576 B
  wa_attn_kernel<<<dim3(NBLK), dim3(256), smem, stream>>>(
      x, mask, wm, wqkvT, wprojT, biasp, qbs, proj_b, out);
}
